// Fusion_Trans_Block_54623394071056
// MI455X (gfx1250) — matrix operand order repaired and checked by reading, not compiled, not run on hardware
//
#include <hip/hip_runtime.h>
#include <hip/hip_bf16.h>

typedef _Float16 f16x8  __attribute__((ext_vector_type(8)));
typedef _Float16 f16x16 __attribute__((ext_vector_type(16)));
typedef float    f32x8  __attribute__((ext_vector_type(8)));

// GCC-style int4 to exactly match the async-LDS builtin's pointee type
typedef int v4i __attribute__((vector_size(16)));
typedef __attribute__((address_space(1))) v4i gv4i;   // global AS1
typedef __attribute__((address_space(3))) v4i lv4i;   // LDS AS3

#define DIM    768
#define HEADS  12
#define HDIM   64
#define HIDDEN 3072
#define SEQ    1024
#define BATCH  8
#define ROWS   (BATCH * SEQ)   // 8192

static __device__ __forceinline__ f32x8 wmma_f16(f16x16 a, f16x16 b, f32x8 c) {
  // D = A(16x32 f16) * B(32x16 f16) + C(16x16 f32)
  return __builtin_amdgcn_wmma_f32_16x16x32_f16(false, a, false, b, (short)0, c,
                                                false, false);
}

// Async global -> LDS copy of 16 bytes per lane (ASYNCcnt path on gfx1250),
// with a synchronous VGPR-round-trip fallback if the builtin is absent.
#if __has_builtin(__builtin_amdgcn_global_load_async_to_lds_b128)
#define HAVE_ASYNC_LDS 1
#endif

static __device__ __forceinline__ void async_copy16(const _Float16* g, _Float16* l) {
#ifdef HAVE_ASYNC_LDS
  __builtin_amdgcn_global_load_async_to_lds_b128((gv4i*)g, (lv4i*)l, 0, 0);
#else
  *(f16x8*)l = *(const f16x8*)g;
#endif
}

static __device__ __forceinline__ void wait_async0() {
#if __has_builtin(__builtin_amdgcn_s_wait_asynccnt)
  __builtin_amdgcn_s_wait_asynccnt(0);
#else
  asm volatile("s_wait_asynccnt 0" ::: "memory");
#endif
}

// ---------------------------------------------------------------------------
// f32 -> f16 weight conversion
// ---------------------------------------------------------------------------
__global__ __launch_bounds__(256) void cvt_f32_f16_kernel(
    const float* __restrict__ src, _Float16* __restrict__ dst, int n) {
  int i = blockIdx.x * 256 + threadIdx.x;
  if (i < n) dst[i] = (_Float16)src[i];
}

// ---------------------------------------------------------------------------
// Fused (optional concat) + LayerNorm.  One wave per row of 768 (24 f32/lane).
// ---------------------------------------------------------------------------
__global__ __launch_bounds__(256) void ln_kernel(
    const float* __restrict__ src_pre, const float* __restrict__ src_post,
    const float* __restrict__ src_full,
    const float* __restrict__ g, const float* __restrict__ beta,
    _Float16* __restrict__ out, float* __restrict__ xout, int concat) {
  const int lane = threadIdx.x & 31;
  const int wid  = threadIdx.x >> 5;
  const int row  = blockIdx.x * 8 + wid;     // 0..8191

  const float* src;
  if (concat) {
    const int b = row >> 10, seq = row & 1023;
    src = (seq < 512) ? src_pre  + ((size_t)b * 512 + seq) * DIM
                      : src_post + ((size_t)b * 512 + (seq - 512)) * DIM;
  } else {
    src = src_full + (size_t)row * DIM;
  }

  float vals[24];
  float s = 0.f, s2 = 0.f;
#pragma unroll
  for (int i = 0; i < 24; ++i) {
    float v = src[lane + i * 32];
    vals[i] = v; s += v; s2 += v * v;
  }
#pragma unroll
  for (int mk = 1; mk < 32; mk <<= 1) {
    s  += __shfl_xor(s,  mk, 32);
    s2 += __shfl_xor(s2, mk, 32);
  }
  const float mu  = s  * (1.f / DIM);
  const float var = s2 * (1.f / DIM) - mu * mu;
  const float rs  = rsqrtf(var + 1e-5f);

#pragma unroll
  for (int i = 0; i < 24; ++i) {
    const int c = lane + i * 32;
    const float v = vals[i];
    if (concat) xout[(size_t)row * DIM + c] = v;
    out[(size_t)row * DIM + c] = (_Float16)((v - mu) * rs * g[c] + beta[c]);
  }
}

// ---------------------------------------------------------------------------
// Generic f16 WMMA GEMM:  C[M x N] = A[M x K](f16) * B[K x N](f16) (+ bias)
// Block tile 64(M) x 128(N), K-step 32, DOUBLE-BUFFERED LDS:
//   - A tile staged via async global->LDS b128 (ASYNCcnt), issued for tile
//     k+1 before computing tile k.
//   - B tile loaded to VGPRs for tile k+1 before compute, scattered to LDS
//     (transposed, k-pairs packed as b32 stores) after compute.
// 8 waves, each owns a 32x32 subtile (2x2 WMMA tiles).
// Epilogue modes:
//   0: out_f16 = acc + bias
//   1: out_f16 = gelu(acc + bias)              (exact, via erff)
//   2: out_f16 transposed-V store: Vt[b][h][d][n_seq]
//   3: out_f32 = resid + gamma * (acc + bias)  (residual stream)
// ---------------------------------------------------------------------------
__global__ __launch_bounds__(256) void gemm_kernel(
    const _Float16* __restrict__ A, int lda,
    const _Float16* __restrict__ B, int ldb,
    const float* __restrict__ bias, int K, int mode,
    _Float16* __restrict__ outh, float* __restrict__ outf,
    const float* __restrict__ resid, const float* __restrict__ gamma, int ldo) {
  __shared__ __attribute__((aligned(32))) _Float16 As[2][64 * 32];
  __shared__ __attribute__((aligned(32))) _Float16 Bt[2][128 * 32];

  const int t    = threadIdx.x;
  const int lane = t & 31, wid = t >> 5;
  const int m0   = blockIdx.y * 64;
  const int n0   = blockIdx.x * 128;
  const int wrow = (wid >> 2) * 32;          // 0 / 32
  const int wcol = (wid & 3) * 32;           // 0 / 32 / 64 / 96
  const int cn   = lane & 15;                // lane row/col within 16
  const int hi   = lane >> 4;                // which lane-half
  const int ak8  = hi * 16;                  // A-fragment k offset (halves)
  const int bk16 = hi * 16;                  // B-fragment k offset (halves)

  const int ar  = t >> 2,  ach = (t & 3) * 8;       // A tile loader mapping
  const int bpk = t >> 4,  bch = (t & 15) * 8;      // B loader: k-pair, col-chunk

  const _Float16* Abase = A + (size_t)(m0 + ar) * lda + ach;
  _Float16* AsDst = (_Float16*)As[0] + ar * 32 + ach;   // [0]/[1] differ by 2048

  // ---- prologue: stage tile 0 into buffer 0 ----
  async_copy16(Abase + 0, AsDst);
  {
    union { f16x8 v; unsigned short u[8]; } lo, hi2;
    lo.v  = *(const f16x8*)(B + (size_t)(2 * bpk)     * ldb + n0 + bch);
    hi2.v = *(const f16x8*)(B + (size_t)(2 * bpk + 1) * ldb + n0 + bch);
    unsigned int* bt32 = (unsigned int*)Bt[0];
#pragma unroll
    for (int i = 0; i < 8; ++i)
      bt32[(bch + i) * 16 + bpk] = (unsigned int)lo.u[i] |
                                   ((unsigned int)hi2.u[i] << 16);
  }
  wait_async0();
  __syncthreads();

  f32x8 acc[2][2] = {};

  for (int k0 = 0; k0 < K; k0 += 32) {
    const int p    = (k0 >> 5) & 1;
    const int np   = p ^ 1;
    const bool more = (k0 + 32) < K;     // uniform

    // ---- issue next tile's fetches before computing ----
    union { f16x8 v; unsigned short u[8]; } nlo = {}, nhi = {};
    if (more) {
      async_copy16(Abase + (k0 + 32), AsDst + np * 2048);
      nlo.v = *(const f16x8*)(B + (size_t)(k0 + 32 + 2 * bpk)     * ldb + n0 + bch);
      nhi.v = *(const f16x8*)(B + (size_t)(k0 + 32 + 2 * bpk + 1) * ldb + n0 + bch);
    }

    // ---- compute on buffer p ----
    const _Float16* Asp = As[p];
    const _Float16* Btp = Bt[p];
    f16x16 a[2], bf[2];
#pragma unroll
    for (int i = 0; i < 2; ++i) {
      union { f16x16 v; f16x8 h[2]; } u;
      u.h[0] = *(const f16x8*)(Asp + (wrow + i * 16 + cn) * 32 + ak8);
      u.h[1] = *(const f16x8*)(Asp + (wrow + i * 16 + cn) * 32 + ak8 + 8);
      a[i] = u.v;
    }
#pragma unroll
    for (int j = 0; j < 2; ++j)
      bf[j] = *(const f16x16*)(Btp + (wcol + j * 16 + cn) * 32 + bk16);
#pragma unroll
    for (int i = 0; i < 2; ++i)
#pragma unroll
      for (int j = 0; j < 2; ++j)
        acc[i][j] = wmma_f16(a[i], bf[j], acc[i][j]);

    // ---- finish staging next tile ----
    if (more) {
      unsigned int* bt32 = (unsigned int*)Bt[np];
#pragma unroll
      for (int i = 0; i < 8; ++i)
        bt32[(bch + i) * 16 + bpk] = (unsigned int)nlo.u[i] |
                                     ((unsigned int)nhi.u[i] << 16);
      wait_async0();
    }
    __syncthreads();
  }

  // ---- epilogue (mode is scalar-uniform -> EXEC stays full everywhere) ----
#pragma unroll
  for (int i = 0; i < 2; ++i)
#pragma unroll
    for (int j = 0; j < 2; ++j) {
      const int gn = n0 + wcol + j * 16 + cn;
      const float bv = bias[gn];
#pragma unroll
      for (int r = 0; r < 8; ++r) {
        const int gm = m0 + wrow + i * 16 + r + hi * 8;
        float val = acc[i][j][r] + bv;
        if (mode == 0) {
          outh[(size_t)gm * ldo + gn] = (_Float16)val;
        } else if (mode == 1) {
          val = 0.5f * val * (1.0f + erff(val * 0.70710678f));
          outh[(size_t)gm * ldo + gn] = (_Float16)val;
        } else if (mode == 2) {
          const int bb = gm >> 10, seq = gm & 1023;
          const int hh = gn >> 6,  dd  = gn & 63;
          outh[(((size_t)bb * HEADS + hh) * HDIM + dd) * SEQ + seq] = (_Float16)val;
        } else {
          const size_t idx = (size_t)gm * ldo + gn;
          outf[idx] = resid[idx] + gamma[gn] * val;
        }
      }
    }
}

// ---------------------------------------------------------------------------
// Flash attention.  One wave per 16-query tile of one (batch, head).
// Per 32-key step: S (2x 16x16 f32 tiles) via 4 WMMA, mask + streaming
// softmax (cross-lane row reductions with xor 1/2/4/8 stay inside the
// 16-lane halves of the C layout), P transposed to A-layout via a private
// 1KB LDS slab, O accumulated via 4 WMMA against Vt[b][h][d][n].
// Next tile's K rows and mask lines are prefetched (global_prefetch_b8).
// ---------------------------------------------------------------------------
__global__ __launch_bounds__(256) void attn_kernel(
    const _Float16* __restrict__ Q, const _Float16* __restrict__ K,
    const _Float16* __restrict__ Vt, const float* __restrict__ mask,
    _Float16* __restrict__ O) {
  __shared__ __attribute__((aligned(32))) _Float16 pst[8][16 * 32];

  const int lane = threadIdx.x & 31;
  const int wid  = threadIdx.x >> 5;
  const int gw   = blockIdx.x * 8 + wid;      // 0..6143
  const int qt   = gw & 63;
  const int head = (gw >> 6) % HEADS;
  const int b    = gw / (64 * HEADS);
  const int q0   = qt * 16;
  _Float16* P = pst[wid];

  const int cn  = lane & 15;
  const int hi  = lane >> 4;
  const int ak8 = hi * 16;
  const int k16 = hi * 16;

  // Q A-fragments (d = 0..31 and 32..63), rows q0+cn
  const _Float16* qrow = Q + ((size_t)(b * SEQ + q0 + cn)) * DIM + head * HDIM;
  f16x16 qa[2];
#pragma unroll
  for (int t = 0; t < 2; ++t) {
    union { f16x16 v; f16x8 h[2]; } u;
    u.h[0] = *(const f16x8*)(qrow + t * 32 + ak8);
    u.h[1] = *(const f16x8*)(qrow + t * 32 + ak8 + 8);
    qa[t] = u.v;
  }

  f32x8 acc[4] = {};
  float mrun[8], lrun[8];
#pragma unroll
  for (int r = 0; r < 8; ++r) { mrun[r] = -3.0e38f; lrun[r] = 0.f; }

  for (int kt = 0; kt < SEQ / 32; ++kt) {
    const int kbase = kt * 32;

    // prefetch next tile's K rows + mask lines into cache early
    if (kt + 1 < SEQ / 32) {
      __builtin_prefetch(K + ((size_t)(b * SEQ + kbase + 32 + lane)) * DIM +
                             head * HDIM, 0, 0);
      __builtin_prefetch(mask + ((size_t)b * SEQ + q0 + hi * 8) * SEQ +
                             kbase + 32 + cn, 0, 0);
    }

    float sv[2][8];
#pragma unroll
    for (int tsub = 0; tsub < 2; ++tsub) {
      const int key = kbase + tsub * 16 + cn;
      const _Float16* krow = K + ((size_t)(b * SEQ + key)) * DIM + head * HDIM;
      f16x16 kb0 = *(const f16x16*)(krow + k16);        // d = k16 .. k16+15
      f16x16 kb1 = *(const f16x16*)(krow + 32 + k16);   // d = 32+k16 ..
      f32x8 c = {};
      c = wmma_f16(qa[0], kb0, c);
      c = wmma_f16(qa[1], kb1, c);
#pragma unroll
      for (int r = 0; r < 8; ++r) sv[tsub][r] = c[r];
    }
    // scale + mask (C layout: row m = q, col n = key)
#pragma unroll
    for (int r = 0; r < 8; ++r) {
      const int qm = q0 + r + hi * 8;
      const float* mp = mask + ((size_t)b * SEQ + qm) * SEQ + kbase;
      sv[0][r] = sv[0][r] * 0.125f + mp[cn];
      sv[1][r] = sv[1][r] * 0.125f + mp[16 + cn];
    }
    // streaming softmax
    float tmax[8];
#pragma unroll
    for (int r = 0; r < 8; ++r) tmax[r] = fmaxf(sv[0][r], sv[1][r]);
#pragma unroll
    for (int mk = 1; mk < 16; mk <<= 1)
#pragma unroll
      for (int r = 0; r < 8; ++r)
        tmax[r] = fmaxf(tmax[r], __shfl_xor(tmax[r], mk, 32));

    float corr[8], rs[8];
#pragma unroll
    for (int r = 0; r < 8; ++r) {
      const float mn = fmaxf(mrun[r], tmax[r]);
      corr[r]  = __expf(mrun[r] - mn);
      mrun[r]  = mn;
      sv[0][r] = __expf(sv[0][r] - mn);
      sv[1][r] = __expf(sv[1][r] - mn);
      rs[r]    = sv[0][r] + sv[1][r];
    }
#pragma unroll
    for (int mk = 1; mk < 16; mk <<= 1)
#pragma unroll
      for (int r = 0; r < 8; ++r) rs[r] += __shfl_xor(rs[r], mk, 32);
#pragma unroll
    for (int r = 0; r < 8; ++r) lrun[r] = lrun[r] * corr[r] + rs[r];
#pragma unroll
    for (int j = 0; j < 4; ++j)
#pragma unroll
      for (int r = 0; r < 8; ++r) acc[j][r] *= corr[r];

    // transpose P (C layout -> A layout) through per-wave LDS slab
#pragma unroll
    for (int r = 0; r < 8; ++r) {
      const int m = r + hi * 8;
      P[m * 32 + cn]      = (_Float16)sv[0][r];
      P[m * 32 + 16 + cn] = (_Float16)sv[1][r];
    }
    asm volatile("s_wait_dscnt 0" ::: "memory");
    union { f16x16 v; f16x8 h[2]; } up;
    up.h[0] = *(const f16x8*)(P + cn * 32 + ak8);
    up.h[1] = *(const f16x8*)(P + cn * 32 + ak8 + 8);
    const f16x16 pa = up.v;

    // O += P * V   (B-frag b[h] = V[key kb16+h][d n]  -> contiguous Vt row)
#pragma unroll
    for (int j = 0; j < 4; ++j) {
      const _Float16* vrow =
          Vt + (((size_t)b * HEADS + head) * HDIM + j * 16 + cn) * SEQ + kbase + k16;
      f16x16 vb = *(const f16x16*)vrow;
      acc[j] = wmma_f16(pa, vb, acc[j]);
    }
  }

  // write O[b, q, head, d] (f16), normalized
#pragma unroll
  for (int j = 0; j < 4; ++j)
#pragma unroll
    for (int r = 0; r < 8; ++r) {
      const int qm = q0 + r + hi * 8;
      const int d  = j * 16 + cn;
      O[((size_t)(b * SEQ + qm)) * DIM + head * HDIM + d] =
          (_Float16)(acc[j][r] / lrun[r]);
    }
}

// ---------------------------------------------------------------------------
// Host-side orchestration
// ---------------------------------------------------------------------------
extern "C" void kernel_launch(void* const* d_in, const int* in_sizes, int n_in,
                              void* d_out, int out_size, void* d_ws, size_t ws_size,
                              hipStream_t stream) {
  (void)in_sizes; (void)n_in; (void)out_size; (void)ws_size;

  const float* x_pre  = (const float*)d_in[0];
  const float* x_post = (const float*)d_in[1];
  const float* mask   = (const float*)d_in[2];
  const float* ln1_g  = (const float*)d_in[3];
  const float* ln1_b  = (const float*)d_in[4];
  const float* Wq = (const float*)d_in[5];   const float* bq = (const float*)d_in[6];
  const float* Wk = (const float*)d_in[7];   const float* bk = (const float*)d_in[8];
  const float* Wv = (const float*)d_in[9];   const float* bv = (const float*)d_in[10];
  const float* Wo = (const float*)d_in[11];  const float* bo = (const float*)d_in[12];
  const float* ln2_g = (const float*)d_in[13]; const float* ln2_b = (const float*)d_in[14];
  const float* W1 = (const float*)d_in[15];  const float* b1 = (const float*)d_in[16];
  const float* W2 = (const float*)d_in[17];  const float* b2 = (const float*)d_in[18];
  const float* g1 = (const float*)d_in[19];  const float* g2 = (const float*)d_in[20];

  char* ws = (char*)d_ws;
  // workspace layout (bytes)
  float*    X   = (float*)(ws + 0);               // 8192x768 f32   (25165824)
  float*    X2  = (float*)(ws + 25165824);        // 8192x768 f32   (25165824)
  _Float16* H   = (_Float16*)(ws + 50331648);     // 8192x768 f16   (12582912)
  _Float16* Qh  = (_Float16*)(ws + 62914560);     // 8192x768 f16
  _Float16* Kh  = (_Float16*)(ws + 75497472);     // 8192x768 f16
  _Float16* Vt  = (_Float16*)(ws + 88080384);     // [B,H,64,1024] f16
  _Float16* Oh  = (_Float16*)(ws + 100663296);    // 8192x768 f16
  _Float16* MID = (_Float16*)(ws + 62914560);     // 8192x3072 f16, aliases Q..O
  _Float16* Wq16 = (_Float16*)(ws + 113246208);
  _Float16* Wk16 = Wq16 + 589824;
  _Float16* Wv16 = Wk16 + 589824;
  _Float16* Wo16 = Wv16 + 589824;
  _Float16* W116 = Wo16 + 589824;
  _Float16* W216 = W116 + 2359296;

  // 1) weight conversion to f16
  cvt_f32_f16_kernel<<<2304, 256, 0, stream>>>(Wq, Wq16, 589824);
  cvt_f32_f16_kernel<<<2304, 256, 0, stream>>>(Wk, Wk16, 589824);
  cvt_f32_f16_kernel<<<2304, 256, 0, stream>>>(Wv, Wv16, 589824);
  cvt_f32_f16_kernel<<<2304, 256, 0, stream>>>(Wo, Wo16, 589824);
  cvt_f32_f16_kernel<<<9216, 256, 0, stream>>>(W1, W116, 2359296);
  cvt_f32_f16_kernel<<<9216, 256, 0, stream>>>(W2, W216, 2359296);

  // 2) concat + LN1  -> X (f32 residual), H (f16)
  ln_kernel<<<ROWS / 8, 256, 0, stream>>>(x_pre, x_post, nullptr,
                                          ln1_g, ln1_b, H, X, 1);

  // 3) Q/K/V projections
  dim3 gemm_grid_768(DIM / 128, ROWS / 64);     // (6, 128)
  gemm_kernel<<<gemm_grid_768, 256, 0, stream>>>(H, DIM, Wq16, DIM, bq, DIM, 0,
                                                 Qh, nullptr, nullptr, nullptr, DIM);
  gemm_kernel<<<gemm_grid_768, 256, 0, stream>>>(H, DIM, Wk16, DIM, bk, DIM, 0,
                                                 Kh, nullptr, nullptr, nullptr, DIM);
  gemm_kernel<<<gemm_grid_768, 256, 0, stream>>>(H, DIM, Wv16, DIM, bv, DIM, 2,
                                                 Vt, nullptr, nullptr, nullptr, DIM);

  // 4) flash attention -> Oh
  attn_kernel<<<(BATCH * HEADS * (SEQ / 16)) / 8, 256, 0, stream>>>(Qh, Kh, Vt,
                                                                    mask, Oh);

  // 5) output projection + residual:  X2 = X + g1 * (Oh @ Wo + bo)
  gemm_kernel<<<gemm_grid_768, 256, 0, stream>>>(Oh, DIM, Wo16, DIM, bo, DIM, 3,
                                                 nullptr, X2, X, g1, DIM);

  // 6) LN2 -> H
  ln_kernel<<<ROWS / 8, 256, 0, stream>>>(nullptr, nullptr, X2,
                                          ln2_g, ln2_b, H, nullptr, 0);

  // 7) MLP fc1 + GELU -> MID
  dim3 gemm_grid_3072(HIDDEN / 128, ROWS / 64);  // (24, 128)
  gemm_kernel<<<gemm_grid_3072, 256, 0, stream>>>(H, DIM, W116, HIDDEN, b1, DIM, 1,
                                                  MID, nullptr, nullptr, nullptr, HIDDEN);

  // 8) MLP fc2 + residual:  out = X2 + g2 * (MID @ W2 + b2)
  gemm_kernel<<<gemm_grid_768, 256, 0, stream>>>(MID, HIDDEN, W216, DIM, b2, HIDDEN, 3,
                                                 nullptr, (float*)d_out, X2, g2, DIM);
}
